// KernelLearningGuide_75935021793751
// MI455X (gfx1250) — matrix operand order repaired and checked by reading, not compiled, not run on hardware
//
#include <hip/hip_runtime.h>
#include <hip/hip_bf16.h>

typedef __attribute__((ext_vector_type(16))) __bf16 v16bf;
typedef __attribute__((ext_vector_type(8)))  __bf16 v8bf;
typedef __attribute__((ext_vector_type(8)))  float  v8f;

#define HW   16384
#define IMW  128
#define HP   130          // padded image dim (halo of 1)
#define HP2  (HP * HP)    // 16900
#define CI   128
#define CG   64
#define CC   (CI + CG)    // 192 concat channels
#define NB   4
#define EPS  1e-5f

__device__ __forceinline__ v8f wmma_bf16(v16bf a, v16bf b, v8f c) {
  return __builtin_amdgcn_wmma_f32_16x16x32_bf16(false, a, false, b, (short)0, c, false, false);
}

// A fragment from bf16 weights [rows][ld], K contiguous; optional row stride.
// lane L: row = row0 + (L%16)*rstride; elem j: K = k0 + 16*(L/16) + j
__device__ __forceinline__ v16bf load_a_bf16_s(const __bf16* __restrict__ W, int ld,
                                               int row0, int rstride, int k0, int lane) {
  int mrow = lane & 15, hi = lane >> 4;
  const __bf16* p = W + (size_t)(row0 + mrow * rstride) * ld + k0 + hi * 16;
  v16bf a;
#pragma unroll
  for (int j = 0; j < 8; ++j) a[j] = p[j];
#pragma unroll
  for (int j = 0; j < 8; ++j) a[8 + j] = p[8 + j];
  return a;
}

// B fragment from pixel-major bf16 tensor with leading dim ldc:
// lane L: col n = L%16; elem j: K = k0 + 16*(L/16) + j  -> one 32B contiguous load.
__device__ __forceinline__ v16bf load_b_pixld(const __bf16* __restrict__ src, int ldc,
                                              int pix, int k0, int lane) {
  int hi = lane >> 4;
  return *(const v16bf*)(src + (size_t)pix * ldc + k0 + hi * 16);
}

// store 8 consecutive channels at one pixel as one b128 store
__device__ __forceinline__ void store8_bf16(__bf16* __restrict__ dst, const float* vals) {
  v8bf o;
#pragma unroll
  for (int v = 0; v < 8; ++v) o[v] = (__bf16)vals[v];
  *(v8bf*)dst = o;
}

// ============ bulk zero (16B chunks) ============
__global__ void k_zero16(uint4* __restrict__ p, long n16) {
  long i = (long)blockIdx.x * 256 + threadIdx.x;
  if (i < n16) p[i] = make_uint4(0u, 0u, 0u, 0u);
}

__global__ void k_zero(float* p, int n) {
  int i = blockIdx.x * 256 + threadIdx.x;
  if (i < n) p[i] = 0.f;
}

// ============ f32 -> bf16 elementwise (same layout) ============
__global__ void k_cvt_bf16(const float* __restrict__ src, __bf16* __restrict__ dst, int n) {
  int i = blockIdx.x * 256 + threadIdx.x;
  if (i < n) dst[i] = (__bf16)src[i];
}

// ============ repack conv weights OIHW f32 -> [O][t*Cin+cin] bf16 (tap-major K) ============
__global__ void k_repack(const float* __restrict__ w, __bf16* __restrict__ out,
                         int Cin, int total) {
  int i = blockIdx.x * 256 + threadIdx.x;
  if (i >= total) return;
  int K = Cin * 9;
  int o = i / K;
  int r = i - o * K;
  int t = r / Cin;
  int cin = r - t * Cin;
  out[i] = (__bf16)w[((size_t)o * Cin + cin) * 9 + t];
}

// ============ build zero-padded pixel-major bf16 concat: [B][130][130][192] ============
__global__ __launch_bounds__(256) void k_fill_wcat(
    const float* __restrict__ in, const float* __restrict__ gw, __bf16* __restrict__ dst)
{
  __shared__ __bf16 tile[16 * CC];
  int blk = blockIdx.x;            // NB * 1024
  int b = blk >> 10;
  int pt = blk & 1023;
  int p0 = pt * 16;
  int y = p0 >> 7, x0 = p0 & (IMW - 1);
  const float* inb = in + (size_t)b * CI * HW + p0;
  const float* gwb = gw + (size_t)b * CG * HW + p0;
  for (int e = threadIdx.x; e < CC * 16; e += 256) {
    int c = e >> 4, p = e & 15;   // coalesced f32 reads along p
    float v = (c < CI) ? inb[(size_t)c * HW + p] : gwb[(size_t)(c - CI) * HW + p];
    tile[p * CC + c] = (__bf16)v;
  }
  __syncthreads();
  __bf16* dp = dst + ((size_t)b * HP2 + (size_t)(y + 1) * HP + x0 + 1) * CC;
  for (int e = threadIdx.x; e < CC * 16; e += 256) {
    int p = e / CC, c = e - p * CC;  // coalesced bf16 writes along c
    dp[(size_t)p * CC + c] = tile[e];
  }
}

// ============ conv11 + conv21, 2-pixel-subtile register blocking ============
// wave = (b, cg, 32-pixel tile): 4 accumulators, A pair shared across 2 B tiles
__global__ __launch_bounds__(256) void k_conv_dual(
    const __bf16* __restrict__ wcat,
    const __bf16* __restrict__ w11r, const float* __restrict__ b11,
    const __bf16* __restrict__ w21r, const float* __restrict__ b21,
    __bf16* __restrict__ w11o, float* __restrict__ pooled)
{
  int wave = blockIdx.x * (blockDim.x >> 5) + (threadIdx.x >> 5);
  int lane = threadIdx.x & 31;
  int pt = wave & 511;             // 512 tiles of 32 pixels
  int cg = (wave >> 9) & 7;
  int b  = wave >> 12;
  int p0 = pt * 32;
  int y = p0 >> 7, x0 = p0 & (IMW - 1);
  int c0 = cg * 16;
  int n = lane & 15, hi = lane >> 4;
  const __bf16* wc = wcat + (size_t)b * HP2 * CC;
  const int KT = CC * 9;  // 1728, tap-major
  v8f a10 = {}, a11v = {}, a20 = {}, a21v = {};
#pragma unroll
  for (int t = 0; t < 9; ++t) {
    const int dy = t / 3, dx = t - 3 * (t / 3);
    int pix = (y + dy) * HP + x0 + n + dx;   // halo-adjusted, always in range
#pragma unroll
    for (int cs = 0; cs < 6; ++cs) {
      const int cc = cs * 32;
      v16bf bf0 = load_b_pixld(wc, CC, pix, cc, lane);
      v16bf bf1 = load_b_pixld(wc, CC, pix + 16, cc, lane);
      int k0 = t * CC + cc;
      v16bf a1 = load_a_bf16_s(w11r, KT, c0, 1, k0, lane);
      v16bf a2 = load_a_bf16_s(w21r, KT, c0, 1, k0, lane);
      a10  = wmma_bf16(a1, bf0, a10);
      a11v = wmma_bf16(a1, bf1, a11v);
      a20  = wmma_bf16(a2, bf0, a20);
      a21v = wmma_bf16(a2, bf1, a21v);
    }
  }
  float r0[8], r1[8];
#pragma unroll
  for (int v = 0; v < 8; ++v) {
    int c = c0 + v + 8 * hi;
    float bb = b11[c];
    r0[v] = fmaxf(a10[v] + bb, 0.f);
    r1[v] = fmaxf(a11v[v] + bb, 0.f);
    float b2 = b21[c];
    float r2 = fmaxf(a20[v] + b2, 0.f) + fmaxf(a21v[v] + b2, 0.f);
#pragma unroll
    for (int s = 1; s < 16; s <<= 1) r2 += __shfl_xor(r2, s, 16);
    if (n == 0) atomicAdd(&pooled[b * CI + c], r2);
  }
  store8_bf16(w11o + ((size_t)b * HW + p0 + n) * CI + c0 + 8 * hi, r0);
  store8_bf16(w11o + ((size_t)b * HW + p0 + 16 + n) * CI + c0 + 8 * hi, r1);
}

// ============ w22o[b][o] = w22[o,:].pooled_mean + b22 (bf16 out) ============
__global__ void k_w22o(const float* __restrict__ w22, const float* __restrict__ b22,
                       const float* __restrict__ pooled, __bf16* __restrict__ w22o)
{
  int idx = blockIdx.x * 256 + threadIdx.x;   // NB*CI*CI = 65536
  int b = idx >> 14, o = idx & 16383;
  const float* pm = pooled + b * CI;
  const float* wr = w22 + (size_t)o * CI;
  float s = b22[o];
#pragma unroll 4
  for (int c = 0; c < CI; ++c) s += wr[c] * (pm[c] * (1.f / (float)HW));
  w22o[idx] = (__bf16)s;
}

// ============ fused conv12 + guided depthwise + BN1 + ReLU ============
__global__ __launch_bounds__(256) void k_guided_local(
    const float* __restrict__ in, const __bf16* __restrict__ w12b,
    const float* __restrict__ b12, const __bf16* __restrict__ w11o,
    const float* __restrict__ g1, const float* __restrict__ be1,
    const float* __restrict__ m1, const float* __restrict__ v1,
    __bf16* __restrict__ localo)
{
  int wave = blockIdx.x * (blockDim.x >> 5) + (threadIdx.x >> 5);
  int lane = threadIdx.x & 31;
  int pt = wave & 1023;
  int cg = (wave >> 10) & 7;
  int b  = wave >> 13;
  int p0 = pt * 16;
  int y = p0 >> 7, x0 = p0 & (IMW - 1);
  int c0 = cg * 16;
  int n = lane & 15, hi = lane >> 4;
  const __bf16* w11ob = w11o + (size_t)b * HW * CI;
  v8f acc[9];
#pragma unroll
  for (int t = 0; t < 9; ++t) acc[t] = (v8f){};
#pragma unroll
  for (int k0 = 0; k0 < CI; k0 += 32) {
    v16bf bf = load_b_pixld(w11ob, CI, p0 + n, k0, lane);
#pragma unroll
    for (int t = 0; t < 9; ++t) {
      // w12 row for (channel c0+mrow, tap t) = (c0+mrow)*9 + t, row stride 9, ld=128
      v16bf a = load_a_bf16_s(w12b, CI, c0 * 9 + t, 9, k0, lane);
      acc[t] = wmma_bf16(a, bf, acc[t]);
    }
  }
  float outv[8];
#pragma unroll
  for (int v = 0; v < 8; ++v) {
    int c = c0 + v + 8 * hi;
    const float* inc = in + ((size_t)b * CI + c) * HW;
    float s = 0.f;
#pragma unroll
    for (int t = 0; t < 9; ++t) {
      int dy = t / 3, dx = t - 3 * (t / 3);
      int yy = y + dy - 1, xx = x0 + n + dx - 1;
      bool ok = ((unsigned)yy < (unsigned)IMW) && ((unsigned)xx < (unsigned)IMW);
      float pv = inc[min(max(yy, 0), IMW - 1) * IMW + min(max(xx, 0), IMW - 1)];
      s += (acc[t][v] + b12[c * 9 + t]) * (ok ? pv : 0.f);
    }
    float sc = g1[c] * rsqrtf(v1[c] + EPS);
    outv[v] = fmaxf((s - m1[c]) * sc + be1[c], 0.f);
  }
  store8_bf16(localo + ((size_t)b * HW + p0 + n) * CI + c0 + 8 * hi, outv);
}

// ============ bmm (w22o @ local) + BN2 + ReLU -> interior of padded out2 ============
__global__ __launch_bounds__(256) void k_bmm(
    const __bf16* __restrict__ w22o, const __bf16* __restrict__ localo,
    const float* __restrict__ g2, const float* __restrict__ be2,
    const float* __restrict__ m2, const float* __restrict__ v2,
    __bf16* __restrict__ out2h)
{
  int wave = blockIdx.x * (blockDim.x >> 5) + (threadIdx.x >> 5);
  int lane = threadIdx.x & 31;
  int pt = wave & 1023;
  int cg = (wave >> 10) & 7;
  int b  = wave >> 13;
  int p0 = pt * 16;
  int y = p0 >> 7, x0 = p0 & (IMW - 1);
  int n = lane & 15, hi = lane >> 4;
  const __bf16* A   = w22o + (size_t)b * CI * CI;
  const __bf16* Bsb = localo + (size_t)b * HW * CI;
  v8f acc = {};
#pragma unroll
  for (int k0 = 0; k0 < CI; k0 += 32) {
    v16bf bf = load_b_pixld(Bsb, CI, p0 + n, k0, lane);
    v16bf a  = load_a_bf16_s(A, CI, cg * 16, 1, k0, lane);
    acc = wmma_bf16(a, bf, acc);
  }
  float outv[8];
#pragma unroll
  for (int v = 0; v < 8; ++v) {
    int c = cg * 16 + v + 8 * hi;
    float sc = g2[c] * rsqrtf(v2[c] + EPS);
    outv[v] = fmaxf((acc[v] - m2[c]) * sc + be2[c], 0.f);
  }
  size_t pixh = (size_t)b * HP2 + (size_t)(y + 1) * HP + x0 + n + 1;
  store8_bf16(out2h + pixh * CI + cg * 16 + 8 * hi, outv);
}

// ============ conv3, 2x2 register blocking (32 channels x 32 pixels per wave) ============
__global__ __launch_bounds__(256) void k_conv3(
    const __bf16* __restrict__ out2h, const __bf16* __restrict__ w3r,
    const float* __restrict__ g3, const float* __restrict__ be3,
    const float* __restrict__ m3, const float* __restrict__ v3,
    float* __restrict__ dout)
{
  int wave = blockIdx.x * (blockDim.x >> 5) + (threadIdx.x >> 5);
  int lane = threadIdx.x & 31;
  int pt  = wave & 511;            // 512 tiles of 32 pixels
  int cgp = (wave >> 9) & 3;       // 4 pairs of channel groups
  int b   = wave >> 11;
  int p0 = pt * 32;
  int y = p0 >> 7, x0 = p0 & (IMW - 1);
  int c0 = cgp * 32;
  int n = lane & 15, hi = lane >> 4;
  const __bf16* src = out2h + (size_t)b * HP2 * CI;
  const int KT = CI * 9;  // 1152, tap-major
  v8f aA0 = {}, aA1 = {}, aB0 = {}, aB1 = {};
#pragma unroll
  for (int t = 0; t < 9; ++t) {
    const int dy = t / 3, dx = t - 3 * (t / 3);
    int pix = (y + dy) * HP + x0 + n + dx;   // halo-adjusted, always in range
#pragma unroll
    for (int cs = 0; cs < 4; ++cs) {
      const int cc = cs * 32;
      v16bf bf0 = load_b_pixld(src, CI, pix, cc, lane);
      v16bf bf1 = load_b_pixld(src, CI, pix + 16, cc, lane);
      int k0 = t * CI + cc;
      v16bf aA = load_a_bf16_s(w3r, KT, c0, 1, k0, lane);
      v16bf aB = load_a_bf16_s(w3r, KT, c0 + 16, 1, k0, lane);
      aA0 = wmma_bf16(aA, bf0, aA0);
      aA1 = wmma_bf16(aA, bf1, aA1);
      aB0 = wmma_bf16(aB, bf0, aB0);
      aB1 = wmma_bf16(aB, bf1, aB1);
    }
  }
#pragma unroll
  for (int v = 0; v < 8; ++v) {
    int cA = c0 + v + 8 * hi;
    int cB = cA + 16;
    float scA = g3[cA] * rsqrtf(v3[cA] + EPS);
    float scB = g3[cB] * rsqrtf(v3[cB] + EPS);
    float vA0 = fmaxf((aA0[v] - m3[cA]) * scA + be3[cA], 0.f);
    float vA1 = fmaxf((aA1[v] - m3[cA]) * scA + be3[cA], 0.f);
    float vB0 = fmaxf((aB0[v] - m3[cB]) * scB + be3[cB], 0.f);
    float vB1 = fmaxf((aB1[v] - m3[cB]) * scB + be3[cB], 0.f);
    dout[((size_t)b * CI + cA) * HW + p0 + n]      = vA0;
    dout[((size_t)b * CI + cA) * HW + p0 + 16 + n] = vA1;
    dout[((size_t)b * CI + cB) * HW + p0 + n]      = vB0;
    dout[((size_t)b * CI + cB) * HW + p0 + 16 + n] = vB1;
  }
}

extern "C" void kernel_launch(void* const* d_in, const int* in_sizes, int n_in,
                              void* d_out, int out_size, void* d_ws, size_t ws_size,
                              hipStream_t stream) {
  const float* in  = (const float*)d_in[0];
  const float* gw  = (const float*)d_in[1];
  const float* w11 = (const float*)d_in[2];
  const float* b11 = (const float*)d_in[3];
  const float* w12 = (const float*)d_in[4];
  const float* b12 = (const float*)d_in[5];
  const float* w21 = (const float*)d_in[6];
  const float* b21 = (const float*)d_in[7];
  const float* w22 = (const float*)d_in[8];
  const float* b22 = (const float*)d_in[9];
  const float* g1  = (const float*)d_in[10];
  const float* be1 = (const float*)d_in[11];
  const float* m1  = (const float*)d_in[12];
  const float* v1  = (const float*)d_in[13];
  const float* g2  = (const float*)d_in[14];
  const float* be2 = (const float*)d_in[15];
  const float* m2  = (const float*)d_in[16];
  const float* v2  = (const float*)d_in[17];
  const float* w3  = (const float*)d_in[18];
  const float* g3  = (const float*)d_in[19];
  const float* be3 = (const float*)d_in[20];
  const float* m3  = (const float*)d_in[21];
  const float* v3  = (const float*)d_in[22];
  float* dout = (float*)d_out;

  // workspace carve-up
  char* ws = (char*)d_ws;
  size_t off = 0;
  const size_t szImg   = (size_t)NB * HW * CI * sizeof(__bf16);    // 16 MB pixel-major
  const size_t szWcat  = (size_t)NB * HP2 * CC * sizeof(__bf16);   // ~26 MB padded concat
  const size_t szOut2h = (size_t)NB * HP2 * CI * sizeof(__bf16);   // ~17 MB padded out2
  __bf16* w11o   = (__bf16*)(ws + off); off += szImg;
  __bf16* localo = (__bf16*)(ws + off); off += szImg;
  __bf16* wcat   = (__bf16*)(ws + off); off += szWcat;
  __bf16* out2h  = (__bf16*)(ws + off); off += szOut2h;
  float*  pooled = (float*)(ws + off);  off += 4096;
  __bf16* w22o   = (__bf16*)(ws + off); off += (size_t)NB * CI * CI * 2;
  const int K11 = CC * 9;   // 1728
  const int K3  = CI * 9;   // 1152
  __bf16* w11r = (__bf16*)(ws + off); off += (size_t)CI * K11 * 2;
  __bf16* w21r = (__bf16*)(ws + off); off += (size_t)CI * K11 * 2;
  __bf16* w3r  = (__bf16*)(ws + off); off += (size_t)CI * K3 * 2;
  __bf16* w12b = (__bf16*)(ws + off); off += (size_t)CI * 9 * CI * 2;

  // prep: zero pooled + halo buffers, build padded wcat, repack weights
  k_zero<<<2, 256, 0, stream>>>(pooled, NB * CI);
  {
    long n16 = (long)(szWcat / 16);
    k_zero16<<<(int)((n16 + 255) / 256), 256, 0, stream>>>((uint4*)wcat, n16);
    long m16 = (long)(szOut2h / 16);
    k_zero16<<<(int)((m16 + 255) / 256), 256, 0, stream>>>((uint4*)out2h, m16);
  }
  k_fill_wcat<<<NB * 1024, 256, 0, stream>>>(in, gw, wcat);
  k_repack<<<(CI * K11 + 255) / 256, 256, 0, stream>>>(w11, w11r, CC, CI * K11);
  k_repack<<<(CI * K11 + 255) / 256, 256, 0, stream>>>(w21, w21r, CC, CI * K11);
  k_repack<<<(CI * K3 + 255) / 256, 256, 0, stream>>>(w3, w3r, CI, CI * K3);
  k_cvt_bf16<<<(CI * 9 * CI + 255) / 256, 256, 0, stream>>>(w12, w12b, CI * 9 * CI);

  // main pipeline
  //   conv_dual: NB * 8 cg * 512 (32-pix tiles) = 16384 waves -> 2048 blocks
  k_conv_dual<<<2048, 256, 0, stream>>>(wcat, w11r, b11, w21r, b21, w11o, pooled);
  k_w22o<<<NB * CI * CI / 256, 256, 0, stream>>>(w22, b22, pooled, w22o);
  //   guided_local / bmm: NB * 8 cg * 1024 (16-pix tiles) = 32768 waves -> 4096 blocks
  k_guided_local<<<4096, 256, 0, stream>>>(in, w12b, b12, w11o, g1, be1, m1, v1, localo);
  k_bmm<<<4096, 256, 0, stream>>>(w22o, localo, g2, be2, m2, v2, out2h);
  //   conv3: NB * 4 cg-pairs * 512 (32-pix tiles) = 8192 waves -> 1024 blocks
  k_conv3<<<1024, 256, 0, stream>>>(out2h, w3r, g3, be3, m3, v3, dout);
}
